// GNNPlusHetero_87419764343139
// MI455X (gfx1250) — matrix operand order repaired and checked by reading, not compiled, not run on hardware
//
#include <hip/hip_runtime.h>
#include <hip/hip_bf16.h>

#define NC 100000
#define NW 500
#define NE 600000
#define NEW 25000
#define H 128
#define L 3
#define DFF 512
#define CF 32
#define WF 16

typedef __bf16 bf16_t;
typedef __attribute__((ext_vector_type(16))) __bf16 v16bf;
typedef __attribute__((ext_vector_type(8)))  __bf16 v8bf;
typedef __attribute__((ext_vector_type(8)))  float  v8f;

static __device__ __forceinline__ bf16_t f2bf(float f) {
  unsigned u = __float_as_uint(f);
  unsigned r = (u + 0x7FFFu + ((u >> 16) & 1u)) >> 16;
  unsigned short h = (unsigned short)r;
  return __builtin_bit_cast(bf16_t, h);
}
static __device__ __forceinline__ unsigned fenc(float f) {
  unsigned u = __float_as_uint(f);
  return (u & 0x80000000u) ? ~u : (u | 0x80000000u);
}
static __device__ __forceinline__ float fdec(unsigned e) {
  unsigned u = (e & 0x80000000u) ? (e & 0x7FFFFFFFu) : ~e;
  return __uint_as_float(u);
}

// ---------------- generic small kernels ----------------

__global__ void zero_kernel(float* __restrict__ p, int n) {
  int i = blockIdx.x * blockDim.x + threadIdx.x;
  if (i < n) p[i] = 0.0f;
}

__global__ void cvt_bf16_kernel(const float* __restrict__ in, bf16_t* __restrict__ out, int n) {
  int i = blockIdx.x * blockDim.x + threadIdx.x;
  if (i < n) out[i] = f2bf(in[i]);
}

// dst[n*K+k] = bf16(src[k*N+n])   (transpose + cvt of a [K,N] weight to [N,K])
__global__ void cvt_t_kernel(const float* __restrict__ src, bf16_t* __restrict__ dst, int K, int N) {
  int i = blockIdx.x * blockDim.x + threadIdx.x;
  if (i >= N * K) return;
  int n = i / K, k = i - n * K;
  dst[i] = f2bf(src[(size_t)k * N + n]);
}

// C[M,N] = (A[M,K] @ B[K,N] + bias) optionally relu  (all f32, naive)
__global__ void naive_mm_kernel(const float* __restrict__ A, const float* __restrict__ B,
                                const float* __restrict__ bias, float* __restrict__ C,
                                int M, int N, int K, int relu) {
  int i = blockIdx.x * blockDim.x + threadIdx.x;
  if (i >= M * N) return;
  int m = i / N, n = i - m * N;
  float acc = bias ? bias[n] : 0.0f;
  const float* a = A + (size_t)m * K;
  for (int k = 0; k < K; ++k) acc += a[k] * B[(size_t)k * N + n];
  if (relu && acc < 0.0f) acc = 0.0f;
  C[i] = acc;
}

// per-layer edge-attention constants: v = gWe[l] @ gae[l];  k1 = Wee.v, k0 = bee.v
__global__ void kconst_kernel(const float* __restrict__ gWe, const float* __restrict__ gae,
                              const float* __restrict__ Wee, const float* __restrict__ bee,
                              float* __restrict__ kc) {
  int l = blockIdx.x, t = threadIdx.x;           // 128 threads
  const float* Wl = gWe + (size_t)l * H * H;
  const float* al = gae + (size_t)l * H;
  float v = 0.0f;
  for (int c = 0; c < H; ++c) v += Wl[(size_t)t * H + c] * al[c];
  __shared__ float s1[H], s0[H];
  s1[t] = Wee[t] * v;
  s0[t] = bee[t] * v;
  __syncthreads();
  for (int o = 64; o > 0; o >>= 1) {
    if (t < o) { s1[t] += s1[t + o]; s0[t] += s0[t + o]; }
    __syncthreads();
  }
  if (t == 0) { kc[l * 2] = s1[0]; kc[l * 2 + 1] = s0[0]; }
}

// ---------------- LayerNorm (wave per row) -> bf16 out ----------------

__global__ void ln_bf16_kernel(const float* __restrict__ h, const float* __restrict__ g,
                               const float* __restrict__ b, bf16_t* __restrict__ out, int nrows) {
  int w = (blockIdx.x * blockDim.x + threadIdx.x) >> 5;
  int lane = threadIdx.x & 31;
  if (w >= nrows) return;
  const float* row = h + (size_t)w * H;
  float x0 = row[lane], x1 = row[lane + 32], x2 = row[lane + 64], x3 = row[lane + 96];
  float s = x0 + x1 + x2 + x3;
  for (int o = 16; o > 0; o >>= 1) s += __shfl_xor(s, o, 32);
  float mu = s * (1.0f / 128.0f);
  float d0 = x0 - mu, d1 = x1 - mu, d2 = x2 - mu, d3 = x3 - mu;
  float v = d0 * d0 + d1 * d1 + d2 * d2 + d3 * d3;
  for (int o = 16; o > 0; o >>= 1) v += __shfl_xor(v, o, 32);
  float rs = rsqrtf(v * (1.0f / 128.0f) + 1e-5f);
  bf16_t* orow = out + (size_t)w * H;
  orow[lane]      = f2bf(d0 * rs * g[lane]      + b[lane]);
  orow[lane + 32] = f2bf(d1 * rs * g[lane + 32] + b[lane + 32]);
  orow[lane + 64] = f2bf(d2 * rs * g[lane + 64] + b[lane + 64]);
  orow[lane + 96] = f2bf(d3 * rs * g[lane + 96] + b[lane + 96]);
}

// ---------------- WMMA bf16 GEMM:  C[M,N] = A[M,K] @ Bt[N,K]^T ----------------
// MODE 0: C_f32 = acc (+bias)   MODE 1: C_bf16 = relu(acc+bias)   MODE 2: C_f32 += acc + bias

template <int MODE>
__global__ void wmma_gemm_kernel(const bf16_t* __restrict__ A, const bf16_t* __restrict__ Bt,
                                 const float* __restrict__ bias, float* __restrict__ Cf,
                                 bf16_t* __restrict__ Cb, int M, int N, int K) {
  int wave = blockIdx.x * (blockDim.x >> 5) + (threadIdx.x >> 5);
  int lane = threadIdx.x & 31;
  int tilesN = N >> 4;
  int tot = (M >> 4) * tilesN;
  if (wave >= tot) return;                       // wave-uniform: EXEC stays all-ones
  int tm = wave / tilesN, tn = wave - tm * tilesN;
  int half = lane >> 4;
  int mn = lane & 15;
  // A frag: lane holds row (tm*16+mn), contiguous K runs [8*half..+7] and [16+8*half..+7]
  const bf16_t* arow = A + (size_t)(tm * 16 + mn) * K + half * 8;
  // B frag (from transposed weights): lane holds col (tn*16+mn), contiguous K runs [8*half..+7] and [16+8*half..+7]
  const bf16_t* brow = Bt + (size_t)(tn * 16 + mn) * K + half * 8;
  v8f acc = {};
  for (int kk = 0; kk < K; kk += 32) {
    v16bf a, bmat;
    *((v8bf*)&a)        = *(const v8bf*)(arow + kk);
    *(((v8bf*)&a) + 1)  = *(const v8bf*)(arow + kk + 16);
    *((v8bf*)&bmat)     = *(const v8bf*)(brow + kk);
    *(((v8bf*)&bmat) + 1) = *(const v8bf*)(brow + kk + 16);
    acc = __builtin_amdgcn_wmma_f32_16x16x32_bf16(false, a, false, bmat,
                                                  (short)0, acc, false, false);
  }
  int n0 = tn * 16 + mn;
  float bi = bias ? bias[n0] : 0.0f;
  int rbase = tm * 16 + half * 8;                // lane<16 -> rows 0..7, lane>=16 -> rows 8..15
#pragma unroll
  for (int r = 0; r < 8; ++r) {
    float v = acc[r] + bi;
    size_t idx = (size_t)(rbase + r) * N + n0;
    if (MODE == 0) Cf[idx] = v;
    else if (MODE == 1) Cb[idx] = f2bf(v > 0.0f ? v : 0.0f);
    else Cf[idx] += v;
  }
}

// ---------------- attention helpers ----------------

// wave per row: o1[r] = x[r].v1 ; o2[r] = x[r].v2
__global__ void dot2_kernel(const float* __restrict__ x, const float* __restrict__ v1,
                            const float* __restrict__ v2, float* __restrict__ o1,
                            float* __restrict__ o2, int nrows) {
  int w = (blockIdx.x * blockDim.x + threadIdx.x) >> 5;
  int lane = threadIdx.x & 31;
  if (w >= nrows) return;
  const float* row = x + (size_t)w * H;
  float s1 = 0.0f, s2 = 0.0f;
  for (int j = lane; j < H; j += 32) {
    float v = row[j];
    s1 += v * v1[j];
    s2 += v * v2[j];
  }
  for (int o = 16; o > 0; o >>= 1) { s1 += __shfl_xor(s1, o, 32); s2 += __shfl_xor(s2, o, 32); }
  if (lane == 0) { o1[w] = s1; o2[w] = s2; }
}

// pass A: raw alpha + segment max via ordered-uint atomicMax
__global__ void edge_alpha_kernel(const int* __restrict__ src, const int* __restrict__ dst,
                                  const float* __restrict__ attr, const float* __restrict__ as,
                                  const float* __restrict__ ad, const float* __restrict__ kc2,
                                  float* __restrict__ ae, unsigned* __restrict__ mu, int nE) {
  int e = blockIdx.x * blockDim.x + threadIdx.x;
  if (e >= nE) return;
  int d = dst[e];
  float a = as[src[e]] + ad[d];
  if (attr) a += kc2[0] * attr[e] + kc2[1];
  a = (a < 0.0f) ? 0.2f * a : a;                 // leaky_relu 0.2
  ae[e] = a;
  atomicMax(&mu[d], fenc(a));
}

// pass B: wave per edge; accumulate unnormalized softmax weights and messages
__global__ void edge_accum_kernel(const int* __restrict__ src, const int* __restrict__ dst,
                                  const float* __restrict__ ae, const unsigned* __restrict__ mu,
                                  float* __restrict__ s, const float* __restrict__ hs,
                                  float* __restrict__ msg, int nE) {
  int e = (blockIdx.x * blockDim.x + threadIdx.x) >> 5;
  int lane = threadIdx.x & 31;
  if (e >= nE) return;
  int sn = src[e], dn = dst[e];
  float ex = __expf(ae[e] - fdec(mu[dn]));
  if (lane == 0) atomicAdd(&s[dn], ex);
  const float* hr = hs + (size_t)sn * H;
  float* mr = msg + (size_t)dn * H;
  atomicAdd(&mr[lane],      hr[lane]      * ex);
  atomicAdd(&mr[lane + 32], hr[lane + 32] * ex);
  atomicAdd(&mr[lane + 64], hr[lane + 64] * ex);
  atomicAdd(&mr[lane + 96], hr[lane + 96] * ex);
}

// pass C: h += msg / max(s,!=0) + gbias
__global__ void apply_msg_kernel(float* __restrict__ h, const float* __restrict__ msg,
                                 const float* __restrict__ s, const float* __restrict__ gbias,
                                 int total) {
  int i = blockIdx.x * blockDim.x + threadIdx.x;
  if (i >= total) return;
  int c = i >> 7, j = i & 127;
  float sv = s[c];
  sv = (sv == 0.0f) ? 1.0f : sv;
  h[i] += msg[i] / sv + gbias[j];
}

// ---------------- host side ----------------

#define CDIV(a, b) (((a) + (b)-1) / (b))
#define BS 256

extern "C" void kernel_launch(void* const* d_in, const int* in_sizes, int n_in,
                              void* d_out, int out_size, void* d_ws, size_t ws_size,
                              hipStream_t stream) {
  const float* cell_x = (const float*)d_in[0];
  const float* well_x = (const float*)d_in[1];
  const int*   eidx   = (const int*)d_in[2];
  const float* eattr  = (const float*)d_in[3];
  const int*   c2w_s  = (const int*)d_in[4];
  const int*   c2w_d  = (const int*)d_in[5];
  const float* W_cell = (const float*)d_in[6];
  const float* b_cell = (const float*)d_in[7];
  const float* W_well = (const float*)d_in[8];
  const float* b_well = (const float*)d_in[9];
  const float* W_ee   = (const float*)d_in[10];
  const float* b_ee   = (const float*)d_in[11];
  const float* gW     = (const float*)d_in[12];
  const float* gas    = (const float*)d_in[13];
  const float* gad    = (const float*)d_in[14];
  const float* gae    = (const float*)d_in[15];
  const float* gWe    = (const float*)d_in[16];
  const float* gb     = (const float*)d_in[17];
  const float* n1g    = (const float*)d_in[18];
  const float* n1b    = (const float*)d_in[19];
  const float* fW1    = (const float*)d_in[20];
  const float* fb1    = (const float*)d_in[21];
  const float* fW2    = (const float*)d_in[22];
  const float* fb2    = (const float*)d_in[23];
  const float* n2g    = (const float*)d_in[24];
  const float* n2b    = (const float*)d_in[25];
  const float* wW     = (const float*)d_in[26];
  const float* was    = (const float*)d_in[27];
  const float* wad    = (const float*)d_in[28];
  const float* wb     = (const float*)d_in[29];
  const float* mW1    = (const float*)d_in[30];
  const float* mb1    = (const float*)d_in[31];
  const float* mW2    = (const float*)d_in[32];
  const float* mb2    = (const float*)d_in[33];
  float* out = (float*)d_out;

  const int* src = eidx;
  const int* dst = eidx + NE;

  // workspace layout
  char* base = (char*)d_ws;
  size_t off = 0;
  auto alloc = [&](size_t bytes) -> char* {
    char* p = base + off;
    off += (bytes + 255) & ~(size_t)255;
    return p;
  };
  float*   h     = (float*)alloc((size_t)NC * H * 4);
  bf16_t*  xnb   = (bf16_t*)alloc((size_t)NC * H * 2);
  float*   hs    = (float*)alloc((size_t)NC * H * 4);      // also reused as hc
  bf16_t*  f1b   = (bf16_t*)alloc((size_t)NC * DFF * 2);
  float*   msg   = (float*)alloc((size_t)NC * H * 4);
  float*   asv   = (float*)alloc((size_t)NC * 4);
  float*   adv   = (float*)alloc((size_t)NC * 4);
  unsigned* muv  = (unsigned*)alloc((size_t)NC * 4);
  float*   sv    = (float*)alloc((size_t)NC * 4);
  float*   ae    = (float*)alloc((size_t)NE * 4);
  bf16_t*  gWt   = (bf16_t*)alloc((size_t)L * H * H * 2);
  bf16_t*  fW1t  = (bf16_t*)alloc((size_t)L * DFF * H * 2);
  bf16_t*  fW2t  = (bf16_t*)alloc((size_t)L * H * DFF * 2);
  bf16_t*  wWt   = (bf16_t*)alloc((size_t)H * H * 2);
  float*   kc    = (float*)alloc((size_t)L * 2 * 4);
  float*   hwv   = (float*)alloc((size_t)NW * H * 4);
  float*   hwp   = (float*)alloc((size_t)NW * H * 4);
  float*   wacc  = (float*)alloc((size_t)NW * H * 4);
  float*   swell = (float*)alloc((size_t)NW * 4);
  unsigned* mwell = (unsigned*)alloc((size_t)NW * 4);
  float*   aew   = (float*)alloc((size_t)NEW * 4);
  float*   adw   = (float*)alloc((size_t)NW * 4);
  float*   t1    = (float*)alloc((size_t)NW * H * 4);

  // --- embeddings ---
  naive_mm_kernel<<<CDIV(NC * H, BS), BS, 0, stream>>>(cell_x, W_cell, b_cell, h, NC, H, CF, 0);
  naive_mm_kernel<<<CDIV(NW * H, BS), BS, 0, stream>>>(well_x, W_well, b_well, hwv, NW, H, WF, 0);

  // --- weight transpose+cvt to bf16 [N,K] ---
  for (int l = 0; l < L; ++l) {
    cvt_t_kernel<<<CDIV(H * H, BS), BS, 0, stream>>>(gW + (size_t)l * H * H, gWt + (size_t)l * H * H, H, H);
    cvt_t_kernel<<<CDIV(H * DFF, BS), BS, 0, stream>>>(fW1 + (size_t)l * H * DFF, fW1t + (size_t)l * DFF * H, H, DFF);
    cvt_t_kernel<<<CDIV(DFF * H, BS), BS, 0, stream>>>(fW2 + (size_t)l * DFF * H, fW2t + (size_t)l * H * DFF, DFF, H);
  }
  cvt_t_kernel<<<CDIV(H * H, BS), BS, 0, stream>>>(wW, wWt, H, H);
  kconst_kernel<<<L, H, 0, stream>>>(gWe, gae, W_ee, b_ee, kc);

  // --- GAT layers ---
  for (int l = 0; l < L; ++l) {
    ln_bf16_kernel<<<CDIV(NC * 32, BS), BS, 0, stream>>>(h, n1g + (size_t)l * H, n1b + (size_t)l * H, xnb, NC);
    {
      int tiles = (NC / 16) * (H / 16);
      wmma_gemm_kernel<0><<<CDIV(tiles, BS / 32), BS, 0, stream>>>(
          xnb, gWt + (size_t)l * H * H, nullptr, hs, nullptr, NC, H, H);
    }
    dot2_kernel<<<CDIV(NC * 32, BS), BS, 0, stream>>>(hs, gas + (size_t)l * H, gad + (size_t)l * H, asv, adv, NC);
    zero_kernel<<<CDIV(NC * H, BS), BS, 0, stream>>>(msg, NC * H);
    zero_kernel<<<CDIV(NC, BS), BS, 0, stream>>>(sv, NC);
    zero_kernel<<<CDIV(NC, BS), BS, 0, stream>>>((float*)muv, NC);
    edge_alpha_kernel<<<CDIV(NE, BS), BS, 0, stream>>>(src, dst, eattr, asv, adv, kc + 2 * l, ae, muv, NE);
    edge_accum_kernel<<<CDIV(NE * 32, BS), BS, 0, stream>>>(src, dst, ae, muv, sv, hs, msg, NE);
    apply_msg_kernel<<<CDIV(NC * H, BS), BS, 0, stream>>>(h, msg, sv, gb + (size_t)l * H, NC * H);

    ln_bf16_kernel<<<CDIV(NC * 32, BS), BS, 0, stream>>>(h, n2g + (size_t)l * H, n2b + (size_t)l * H, xnb, NC);
    {
      int tiles = (NC / 16) * (DFF / 16);
      wmma_gemm_kernel<1><<<CDIV(tiles, BS / 32), BS, 0, stream>>>(
          xnb, fW1t + (size_t)l * DFF * H, fb1 + (size_t)l * DFF, nullptr, f1b, NC, DFF, H);
    }
    {
      int tiles = (NC / 16) * (H / 16);
      wmma_gemm_kernel<2><<<CDIV(tiles, BS / 32), BS, 0, stream>>>(
          f1b, fW2t + (size_t)l * H * DFF, fb2 + (size_t)l * H, h, nullptr, NC, H, DFF);
    }
  }

  // --- bipartite cell -> well GAT ---
  cvt_bf16_kernel<<<CDIV(NC * H, BS), BS, 0, stream>>>(h, xnb, NC * H);
  {
    int tiles = (NC / 16) * (H / 16);
    wmma_gemm_kernel<0><<<CDIV(tiles, BS / 32), BS, 0, stream>>>(
        xnb, wWt, nullptr, hs, nullptr, NC, H, H);           // hs = hc
  }
  dot2_kernel<<<CDIV(NC * 32, BS), BS, 0, stream>>>(hs, was, was, asv, adv, NC);  // asv = hc.was
  naive_mm_kernel<<<CDIV(NW * H, BS), BS, 0, stream>>>(hwv, wW, nullptr, hwp, NW, H, H, 0);
  naive_mm_kernel<<<CDIV(NW, BS), BS, 0, stream>>>(hwp, wad, nullptr, adw, NW, 1, H, 0);
  zero_kernel<<<CDIV(NW * H, BS), BS, 0, stream>>>(wacc, NW * H);
  zero_kernel<<<CDIV(NW, BS), BS, 0, stream>>>(swell, NW);
  zero_kernel<<<CDIV(NW, BS), BS, 0, stream>>>((float*)mwell, NW);
  edge_alpha_kernel<<<CDIV(NEW, BS), BS, 0, stream>>>(c2w_s, c2w_d, nullptr, asv, adw, nullptr, aew, mwell, NEW);
  edge_accum_kernel<<<CDIV(NEW * 32, BS), BS, 0, stream>>>(c2w_s, c2w_d, aew, mwell, swell, hs, wacc, NEW);
  zero_kernel<<<CDIV(NW * H, BS), BS, 0, stream>>>(hwp, NW * H);
  apply_msg_kernel<<<CDIV(NW * H, BS), BS, 0, stream>>>(hwp, wacc, swell, wb, NW * H);

  // --- head ---
  naive_mm_kernel<<<CDIV(NW * H, BS), BS, 0, stream>>>(hwp, mW1, mb1, t1, NW, H, H, 1);
  naive_mm_kernel<<<CDIV(NW * 75, BS), BS, 0, stream>>>(t1, mW2, mb2, out, NW, 75, H, 0);

  (void)in_sizes; (void)n_in; (void)out_size; (void)ws_size;
}
